// SAMVisionLayer_9904194584922
// MI455X (gfx1250) — matrix operand order repaired and checked by reading, not compiled, not run on hardware
//
#include <hip/hip_runtime.h>
#include <hip/hip_bf16.h>
#include <math.h>

// ---------------- problem constants ----------------
#define WSZ   14          // window size
#define NH    12          // heads
#define HD    64          // head dim
#define HS    768         // hidden
#define MLP   3072
#define S     196         // tokens per window (14*14)
#define STIL  13          // ceil(196/16)
#define S2    224         // S padded to multiple of 32 for PV K-loop
#define NWIN  100         // 4 imgs * 25 windows
#define MTOK  19600       // NWIN * S   (multiple of 16)
#define MFLAT 16384       // 4*64*64

typedef __attribute__((ext_vector_type(16))) _Float16 v16h;
typedef __attribute__((ext_vector_type(8)))  _Float16 v8h;
typedef __attribute__((ext_vector_type(8)))  float    v8f;

// CDNA5 async global->LDS path (gated: not probe-confirmed on all toolchains)
#if defined(__has_builtin)
#if __has_builtin(__builtin_amdgcn_global_load_async_to_lds_b128) && \
    __has_builtin(__builtin_amdgcn_s_wait_asynccnt)
#define USE_ASYNC_LDS 1
#endif
#endif
#ifndef USE_ASYNC_LDS
#define USE_ASYNC_LDS 0
#endif

// ---------------- small helpers ----------------
__device__ inline v8f zero8() {
  v8f z;
#pragma unroll
  for (int i = 0; i < 8; ++i) z[i] = 0.0f;
  return z;
}

__device__ inline v8f wmma_f16(v16h a, v16h b, v8f c) {
  // emits v_wmma_f32_16x16x32_f16
  return __builtin_amdgcn_wmma_f32_16x16x32_f16(false, a, false, b, (short)0, c,
                                                false, false);
}

// A-fragment (16x32 f16), ISA 7.12.2: lane L -> row = L%16;
// lanes 0-15 hold K = k0+0..7 (v0..3) and k0+16..23 (v4..7),
// lanes 16-31 hold K = k0+8..15 and k0+24..31.
__device__ inline v16h load_fragA(const _Float16* base, int ld, int rlimit,
                                  int r0, int k0) {
  const int lane = threadIdx.x & 31;
  const int row  = r0 + (lane & 15);
  const int half = lane >> 4;
  v16h f;
  if (row < rlimit) {
    const _Float16* p = base + (size_t)row * ld + k0 + half * 8;
    v8h a = *(const v8h*)(p);
    v8h b = *(const v8h*)(p + 16);
#pragma unroll
    for (int i = 0; i < 8; ++i) { f[i] = a[i]; f[i + 8] = b[i]; }
  } else {
#pragma unroll
    for (int i = 0; i < 16; ++i) f[i] = (_Float16)0.0f;
  }
  return f;
}

// B-fragment (32x16 f16) fed from Bt (row-major [N,K], i.e. W transposed):
// lane L -> col n = L%16; lanes 0-15 hold K = k0+0..7 and k0+16..23,
// lanes 16-31 hold K = k0+8..15 and k0+24..31.
__device__ inline v16h load_fragB(const _Float16* bt, int ld, int nlimit,
                                  int n0, int k0) {
  const int lane = threadIdx.x & 31;
  const int row  = n0 + (lane & 15);
  const int half = lane >> 4;
  v16h f;
  if (row < nlimit) {
    const _Float16* p = bt + (size_t)row * ld + k0 + half * 8;
    v8h a = *(const v8h*)(p);
    v8h b = *(const v8h*)(p + 16);
#pragma unroll
    for (int i = 0; i < 8; ++i) { f[i] = a[i]; f[i + 8] = b[i]; }
  } else {
#pragma unroll
    for (int i = 0; i < 16; ++i) f[i] = (_Float16)0.0f;
  }
  return f;
}

#if USE_ASYNC_LDS
// one 16-byte async DMA: global -> LDS (tracked by ASYNCcnt).
// Builtin signature (from hipcc diagnostic): (v4i AS1*, v4i AS3*, imm, imm).
typedef int avi4 __attribute__((vector_size(16)));
typedef avi4 __attribute__((address_space(1)))* g4p;
typedef avi4 __attribute__((address_space(3)))* l4p;
__device__ inline void async_cp16(const _Float16* gsrc, _Float16* lds_dst) {
  __builtin_amdgcn_global_load_async_to_lds_b128((g4p)gsrc, (l4p)lds_dst, 0, 0);
}
#endif

// ---------------- weight convert (fp32 [K,N] -> f16 transposed [N,K]) -------
__global__ void convert_wt_kernel(const float* __restrict__ w,
                                  _Float16* __restrict__ wt, int K, int N) {
  long long idx = (long long)blockIdx.x * blockDim.x + threadIdx.x;
  if (idx >= (long long)K * N) return;
  int n = (int)(idx / K);
  int k = (int)(idx % K);
  wt[idx] = (_Float16)w[(size_t)k * N + n];
}

// ---------------- LN1 + window partition -> f16 ----------------
__global__ __launch_bounds__(256)
void ln1_window_kernel(const float* __restrict__ x, const float* __restrict__ g,
                       const float* __restrict__ bta, _Float16* __restrict__ xw) {
  const int t  = blockIdx.x;        // window token id 0..MTOK-1
  const int bw = t / S;
  const int s  = t % S;
  const int wh = s / WSZ, ww = s % WSZ;
  const int b  = bw / 25, wi = bw % 25;
  const int h  = (wi / 5) * WSZ + wh;
  const int w  = (wi % 5) * WSZ + ww;
  const int tid = threadIdx.x;
  _Float16* outp = xw + (size_t)t * HS;
  if (h >= 64 || w >= 64) {                 // padded token -> zeros
    for (int i = tid; i < HS; i += 256) outp[i] = (_Float16)0.0f;
    return;
  }
  const float* xp = x + (((size_t)b * 64 + h) * 64 + w) * HS;
  float v0 = xp[tid], v1 = xp[tid + 256], v2 = xp[tid + 512];
  __shared__ float r1[256], r2[256];
  r1[tid] = v0 + v1 + v2;
  r2[tid] = v0 * v0 + v1 * v1 + v2 * v2;
  __syncthreads();
  for (int st = 128; st > 0; st >>= 1) {
    if (tid < st) { r1[tid] += r1[tid + st]; r2[tid] += r2[tid + st]; }
    __syncthreads();
  }
  const float mu  = r1[0] * (1.0f / HS);
  const float var = r2[0] * (1.0f / HS) - mu * mu;
  const float rs  = rsqrtf(var + 1e-6f);
  outp[tid]       = (_Float16)((v0 - mu) * rs * g[tid]       + bta[tid]);
  outp[tid + 256] = (_Float16)((v1 - mu) * rs * g[tid + 256] + bta[tid + 256]);
  outp[tid + 512] = (_Float16)((v2 - mu) * rs * g[tid + 512] + bta[tid + 512]);
}

// ---------------- LN2 (flat tokens) -> f16 ----------------
__global__ __launch_bounds__(256)
void ln2_kernel(const float* __restrict__ x2, const float* __restrict__ g,
                const float* __restrict__ bta, _Float16* __restrict__ y) {
  const int t = blockIdx.x;
  const int tid = threadIdx.x;
  const float* xp = x2 + (size_t)t * HS;
  _Float16* outp = y + (size_t)t * HS;
  float v0 = xp[tid], v1 = xp[tid + 256], v2 = xp[tid + 512];
  __shared__ float r1[256], r2[256];
  r1[tid] = v0 + v1 + v2;
  r2[tid] = v0 * v0 + v1 * v1 + v2 * v2;
  __syncthreads();
  for (int st = 128; st > 0; st >>= 1) {
    if (tid < st) { r1[tid] += r1[tid + st]; r2[tid] += r2[tid + st]; }
    __syncthreads();
  }
  const float mu  = r1[0] * (1.0f / HS);
  const float var = r2[0] * (1.0f / HS) - mu * mu;
  const float rs  = rsqrtf(var + 1e-6f);
  outp[tid]       = (_Float16)((v0 - mu) * rs * g[tid]       + bta[tid]);
  outp[tid + 256] = (_Float16)((v1 - mu) * rs * g[tid + 256] + bta[tid + 256]);
  outp[tid + 512] = (_Float16)((v2 - mu) * rs * g[tid + 512] + bta[tid + 512]);
}

// ---------------- rel-pos dot products ----------------
__global__ void relpos_kernel(const _Float16* __restrict__ q,
                              const float* __restrict__ rph,
                              const float* __restrict__ rpw,
                              float* __restrict__ relh,
                              float* __restrict__ relw) {
  const long long total = (long long)NWIN * NH * S * WSZ;
  long long idx = (long long)blockIdx.x * blockDim.x + threadIdx.x;
  if (idx >= total) return;
  const int j   = (int)(idx % WSZ);
  const int s   = (int)((idx / WSZ) % S);
  const int bwh = (int)(idx / (WSZ * S));
  const _Float16* qp = q + ((size_t)bwh * S + s) * HD;
  const int qh = s / WSZ, qw = s % WSZ;
  const float* rh = rph + (size_t)(qh - j + WSZ - 1) * HD;
  const float* rw = rpw + (size_t)(qw - j + WSZ - 1) * HD;
  float ah = 0.0f, aw = 0.0f;
#pragma unroll 8
  for (int d = 0; d < HD; ++d) {
    const float qv = (float)qp[d];
    ah += qv * rh[d];
    aw += qv * rw[d];
  }
  relh[idx] = ah;
  relw[idx] = aw;
}

// ---------------- generic WMMA GEMM: C = A[M,K] * Bt[N,K]^T + epilogue ------
// block = 128 threads (4 waves in 2x2); each wave computes a 32x32 tile.
#if USE_ASYNC_LDS
// CDNA5 path: cooperative async-DMA staging into double-buffered LDS;
// ASYNCcnt-gated; WMMA consumes ds_load_b128 fragments.
template <class Epi>
__global__ __launch_bounds__(128)
void gemm_wmma_kernel(const _Float16* __restrict__ A,
                      const _Float16* __restrict__ Bt,
                      int M, int N, int K, Epi epi) {
  __shared__ _Float16 As[2][64][32];
  __shared__ _Float16 Bs[2][64][32];
  const int tid  = threadIdx.x;
  const int wave = tid >> 5;
  const int lane = tid & 31;
  const int half = lane >> 4;
  const int nl   = lane & 15;
  const int m0 = blockIdx.y * 64;
  const int n0 = blockIdx.x * 64;
  const int wm0 = (wave >> 1) * 32;        // wave tile inside block
  const int wn0 = (wave & 1) * 32;

  // issue one 64x32 A-tile + 64x32 B-tile stage (4 async b128 ops per wave)
  auto stage = [&](int k0, int buf) {
#pragma unroll
    for (int c = 0; c < 2; ++c) {
      const int chunk = tid + c * 128;      // 0..255 16B chunks
      const int row = chunk >> 2;           // 0..63
      const int col = (chunk & 3) * 8;      // 0,8,16,24
      int ar = m0 + row; if (ar >= M) ar = M - 1;   // clamp: garbage rows are
      int br = n0 + row; if (br >= N) br = N - 1;   // discarded at store time
      async_cp16(A  + (size_t)ar * K + k0 + col, &As[buf][row][col]);
      async_cp16(Bt + (size_t)br * K + k0 + col, &Bs[buf][row][col]);
    }
  };

  v8f acc00 = zero8(), acc01 = zero8(), acc10 = zero8(), acc11 = zero8();
  const int nk = K / 32;                    // K is always a multiple of 32 here
  stage(0, 0);
  for (int kt = 0; kt < nk; ++kt) {
    const int buf = kt & 1;
    if (kt + 1 < nk) {
      stage((kt + 1) * 32, buf ^ 1);
      __builtin_amdgcn_s_wait_asynccnt(4);  // loads complete in order: tile kt landed
    } else {
      __builtin_amdgcn_s_wait_asynccnt(0);
    }
    __syncthreads();
    v16h a0, a1, b0, b1;
    {
      const _Float16* p = &As[buf][wm0 + nl][half * 8];
      v8h x = *(const v8h*)p, y = *(const v8h*)(p + 16);
#pragma unroll
      for (int i = 0; i < 8; ++i) { a0[i] = x[i]; a0[i + 8] = y[i]; }
      p = &As[buf][wm0 + 16 + nl][half * 8];
      x = *(const v8h*)p; y = *(const v8h*)(p + 16);
#pragma unroll
      for (int i = 0; i < 8; ++i) { a1[i] = x[i]; a1[i + 8] = y[i]; }
      p = &Bs[buf][wn0 + nl][half * 8];
      x = *(const v8h*)p; y = *(const v8h*)(p + 16);
#pragma unroll
      for (int i = 0; i < 8; ++i) { b0[i] = x[i]; b0[i + 8] = y[i]; }
      p = &Bs[buf][wn0 + 16 + nl][half * 8];
      x = *(const v8h*)p; y = *(const v8h*)(p + 16);
#pragma unroll
      for (int i = 0; i < 8; ++i) { b1[i] = x[i]; b1[i + 8] = y[i]; }
    }
    acc00 = wmma_f16(a0, b0, acc00);
    acc01 = wmma_f16(a0, b1, acc01);
    acc10 = wmma_f16(a1, b0, acc10);
    acc11 = wmma_f16(a1, b1, acc11);
    __syncthreads();                        // buf reused by stage kt+2
  }
#pragma unroll
  for (int i = 0; i < 8; ++i) {
    const int r0 = m0 + wm0 + i + 8 * half;
    const int r1 = r0 + 16;
    const int c0 = n0 + wn0 + nl;
    const int c1 = c0 + 16;
    if (r0 < M && c0 < N) epi(acc00[i], r0, c0);
    if (r0 < M && c1 < N) epi(acc01[i], r0, c1);
    if (r1 < M && c0 < N) epi(acc10[i], r1, c0);
    if (r1 < M && c1 < N) epi(acc11[i], r1, c1);
  }
}
#else
// fallback: direct global-load fragments (proven codegen)
template <class Epi>
__global__ __launch_bounds__(128)
void gemm_wmma_kernel(const _Float16* __restrict__ A,
                      const _Float16* __restrict__ Bt,
                      int M, int N, int K, Epi epi) {
  const int wave = threadIdx.x >> 5;
  const int m0 = blockIdx.y * 64 + (wave >> 1) * 32;
  const int n0 = blockIdx.x * 64 + (wave & 1) * 32;
  v8f acc00 = zero8(), acc01 = zero8(), acc10 = zero8(), acc11 = zero8();
  for (int k0 = 0; k0 < K; k0 += 32) {
    if (k0 + 32 < K) {   // hint next K-tile into cache (global_prefetch_b8)
      __builtin_prefetch(A + (size_t)m0 * K + k0 + 32, 0, 1);
      __builtin_prefetch(Bt + (size_t)n0 * K + k0 + 32, 0, 1);
    }
    const v16h a0 = load_fragA(A, K, M, m0,      k0);
    const v16h a1 = load_fragA(A, K, M, m0 + 16, k0);
    const v16h b0 = load_fragB(Bt, K, N, n0,      k0);
    const v16h b1 = load_fragB(Bt, K, N, n0 + 16, k0);
    acc00 = wmma_f16(a0, b0, acc00);
    acc01 = wmma_f16(a0, b1, acc01);
    acc10 = wmma_f16(a1, b0, acc10);
    acc11 = wmma_f16(a1, b1, acc11);
  }
  const int lane = threadIdx.x & 31;
  const int half = lane >> 4;
  const int nl   = lane & 15;
#pragma unroll
  for (int i = 0; i < 8; ++i) {
    const int r0 = m0 + i + 8 * half;       // C layout: M = vgpr + 8*(lane/16)
    const int r1 = r0 + 16;
    const int c0 = n0 + nl;
    const int c1 = c0 + 16;
    if (r0 < M && c0 < N) epi(acc00[i], r0, c0);
    if (r0 < M && c1 < N) epi(acc01[i], r0, c1);
    if (r1 < M && c0 < N) epi(acc10[i], r1, c0);
    if (r1 < M && c1 < N) epi(acc11[i], r1, c1);
  }
}
#endif

// ---------------- GEMM epilogues ----------------
struct EpiQKV {            // scatter into q[bwh][S][HD], k[bwh][S][HD], vT[bwh][HD][S2]
  const float* bias;
  _Float16* q; _Float16* k; _Float16* v;
  __device__ void operator()(float val, int m, int n) const {
    const int bw = m / S, s = m % S;
    const int which = n / HS, c = n % HS;
    const int head = c / HD, d = c % HD;
    const int bwh = bw * NH + head;
    const _Float16 r = (_Float16)(val + bias[n]);
    if (which == 0)      q[((size_t)bwh * S + s) * HD + d] = r;
    else if (which == 1) k[((size_t)bwh * S + s) * HD + d] = r;
    else                 v[((size_t)bwh * HD + d) * S2 + s] = r;   // transposed
  }
};

struct EpiProj {           // window-unpartition + bias + residual -> x2 (fp32)
  const float* bias; const float* x; float* x2;
  __device__ void operator()(float val, int m, int n) const {
    const int bw = m / S, s = m % S;
    const int wh = s / WSZ, ww = s % WSZ;
    const int b = bw / 25, wi = bw % 25;
    const int h = (wi / 5) * WSZ + wh;
    const int w = (wi % 5) * WSZ + ww;
    if (h < 64 && w < 64) {
      const size_t idx = (((size_t)b * 64 + h) * 64 + w) * HS + n;
      x2[idx] = val + bias[n] + x[idx];
    }
  }
};

struct EpiGelu {           // bias + exact GELU -> f16
  const float* bias; _Float16* out;
  __device__ void operator()(float val, int m, int n) const {
    const float v = val + bias[n];
    const float g = 0.5f * v * (1.0f + erff(v * 0.70710678118654752f));
    out[(size_t)m * MLP + n] = (_Float16)g;
  }
};

struct EpiFC2 {            // bias + residual -> final output (fp32)
  const float* bias; const float* x2; float* out;
  __device__ void operator()(float val, int m, int n) const {
    const size_t idx = (size_t)m * HS + n;
    out[idx] = val + bias[n] + x2[idx];
  }
};

// ---------------- attention: one wave per (window-head, 16-query tile) ------
__global__ __launch_bounds__(32)
void attn_kernel(const _Float16* __restrict__ q, const _Float16* __restrict__ k,
                 const _Float16* __restrict__ v, const float* __restrict__ relh,
                 const float* __restrict__ relw, _Float16* __restrict__ out) {
  const int bwh = blockIdx.x / STIL;
  const int mt  = blockIdx.x % STIL;
  const int m0  = mt * 16;
  const _Float16* qb = q + (size_t)bwh * S * HD;
  const _Float16* kb = k + (size_t)bwh * S * HD;
  const _Float16* vb = v + (size_t)bwh * HD * S2;
  const int lane = threadIdx.x & 31;
  const int half = lane >> 4;
  const int nl   = lane & 15;

  // ---- scores = q @ k^T  (13 N-tiles, K = HD = 64) ----
  v8f sc[STIL];
#pragma unroll
  for (int t = 0; t < STIL; ++t) sc[t] = zero8();
#pragma unroll
  for (int kk = 0; kk < HD; kk += 32) {
    const v16h aq = load_fragA(qb, HD, S, m0, kk);
#pragma unroll
    for (int t = 0; t < STIL; ++t) {
      const v16h bk = load_fragB(kb, HD, S, t * 16, kk);
      sc[t] = wmma_f16(aq, bk, sc[t]);
    }
  }

  // ---- scale + rel-pos bias + key mask ----
#pragma unroll
  for (int t = 0; t < STIL; ++t) {
    const int col = t * 16 + nl;                 // key index
#pragma unroll
    for (int i = 0; i < 8; ++i) {
      const int row = m0 + i + 8 * half;         // query index
      float s = sc[t][i] * 0.125f;               // HD^-0.5
      if (row < S && col < S) {
        const int kh = col / WSZ, kw = col % WSZ;
        const size_t rb = ((size_t)bwh * S + row) * WSZ;
        s += relh[rb + kh] + relw[rb + kw];
      } else {
        s = -1e30f;
      }
      sc[t][i] = s;
    }
  }

  // ---- softmax per row (rows live in 16-lane halves; butterfly in-half) ----
  float rinv[8];
#pragma unroll
  for (int i = 0; i < 8; ++i) {
    float mx = -3.4e38f;
#pragma unroll
    for (int t = 0; t < STIL; ++t) mx = fmaxf(mx, sc[t][i]);
    for (int off = 1; off < 16; off <<= 1) mx = fmaxf(mx, __shfl_xor(mx, off, 32));
    float sum = 0.0f;
#pragma unroll
    for (int t = 0; t < STIL; ++t) {
      const float e = __expf(sc[t][i] - mx);
      sc[t][i] = e;
      sum += e;
    }
    for (int off = 1; off < 16; off <<= 1) sum += __shfl_xor(sum, off, 32);
    rinv[i] = 1.0f / sum;
  }

  // ---- bounce P through LDS to reshape into A-fragment layout ----
  __shared__ _Float16 P[16][S2];
#pragma unroll
  for (int t = 0; t < STIL; ++t) {
#pragma unroll
    for (int i = 0; i < 8; ++i)
      P[i + 8 * half][t * 16 + nl] = (_Float16)(sc[t][i] * rinv[i]);
  }
#pragma unroll
  for (int i = 0; i < 8; ++i)                    // zero pad cols 208..223
    P[i + 8 * half][STIL * 16 + nl] = (_Float16)0.0f;
  __syncthreads();

  // ---- O = P @ V  (N = HD = 64 -> 4 tiles, K = S2) ----
  v8f o[4];
#pragma unroll
  for (int t = 0; t < 4; ++t) o[t] = zero8();
  for (int kk = 0; kk < S2; kk += 32) {
    v16h ap;
    {
      const v8h a = *(const v8h*)&P[nl][kk + half * 8];
      const v8h b = *(const v8h*)&P[nl][kk + 16 + half * 8];
#pragma unroll
      for (int i = 0; i < 8; ++i) { ap[i] = a[i]; ap[i + 8] = b[i]; }
    }
#pragma unroll
    for (int t = 0; t < 4; ++t) {
      const v16h bv = load_fragB(vb, S2, HD, t * 16, kk);
      o[t] = wmma_f16(ap, bv, o[t]);
    }
  }

  // ---- store to attn buffer [bw][s][head*64+d] (f16, feeds proj GEMM) ----
  const int bw = bwh / NH, head = bwh % NH;
#pragma unroll
  for (int t = 0; t < 4; ++t) {
#pragma unroll
    for (int i = 0; i < 8; ++i) {
      const int row = m0 + i + 8 * half;
      if (row < S) {
        const int d = t * 16 + nl;
        out[((size_t)bw * S + row) * HS + head * HD + d] = (_Float16)o[t][i];
      }
    }
  }
}

// ---------------- host launcher ----------------
static inline size_t align256(size_t x) { return (x + 255) & ~(size_t)255; }
static inline unsigned cdiv(unsigned a, unsigned b) { return (a + b - 1) / b; }

extern "C" void kernel_launch(void* const* d_in, const int* in_sizes, int n_in,
                              void* d_out, int out_size, void* d_ws, size_t ws_size,
                              hipStream_t stream) {
  const float* x       = (const float*)d_in[0];
  const float* ln1_g   = (const float*)d_in[1];
  const float* ln1_b   = (const float*)d_in[2];
  const float* qkv_w   = (const float*)d_in[3];
  const float* qkv_b   = (const float*)d_in[4];
  const float* proj_w  = (const float*)d_in[5];
  const float* proj_b  = (const float*)d_in[6];
  const float* rph     = (const float*)d_in[7];
  const float* rpw     = (const float*)d_in[8];
  const float* ln2_g   = (const float*)d_in[9];
  const float* ln2_b   = (const float*)d_in[10];
  const float* fc1_w   = (const float*)d_in[11];
  const float* fc1_b   = (const float*)d_in[12];
  const float* fc2_w   = (const float*)d_in[13];
  const float* fc2_b   = (const float*)d_in[14];
  float* out = (float*)d_out;

  // workspace layout (f16 staging + fp32 x2); h1 / y alias dead regions
  char* ws = (char*)d_ws;
  size_t off = 0;
  _Float16* qkv_wt = (_Float16*)(ws + off); off = align256(off + (size_t)3 * HS * HS * 2);
  _Float16* proj_wt= (_Float16*)(ws + off); off = align256(off + (size_t)HS * HS * 2);
  _Float16* fc1_wt = (_Float16*)(ws + off); off = align256(off + (size_t)HS * MLP * 2);
  _Float16* fc2_wt = (_Float16*)(ws + off); off = align256(off + (size_t)MLP * HS * 2);
  _Float16* xw     = (_Float16*)(ws + off); size_t o_xw = off;
                                            off = align256(off + (size_t)MTOK * HS * 2);
  _Float16* qh     = (_Float16*)(ws + off); size_t o_q = off;
                                            off = align256(off + (size_t)NWIN * NH * S * HD * 2);
  _Float16* kh     = (_Float16*)(ws + off); off = align256(off + (size_t)NWIN * NH * S * HD * 2);
  _Float16* vh     = (_Float16*)(ws + off); off = align256(off + (size_t)NWIN * NH * HD * S2 * 2);
  float*    relh   = (float*)   (ws + off); off = align256(off + (size_t)NWIN * NH * S * WSZ * 4);
  float*    relw   = (float*)   (ws + off); off = align256(off + (size_t)NWIN * NH * S * WSZ * 4);
  _Float16* attn_h = (_Float16*)(ws + off); off = align256(off + (size_t)MTOK * HS * 2);
  float*    x2     = (float*)   (ws + off); off = align256(off + (size_t)MFLAT * HS * 4);
  _Float16* y      = (_Float16*)(ws + o_xw);   // LN2 output reuses xw region
  _Float16* h1     = (_Float16*)(ws + o_q);    // MLP hidden reuses q/k/v/rel region
  (void)ws_size; (void)n_in; (void)in_sizes; (void)out_size;

  // 1) weight convert+transpose to f16 (L2-resident thereafter)
  {
    long long n;
    n = (long long)HS * 3 * HS;
    convert_wt_kernel<<<cdiv((unsigned)n, 256), 256, 0, stream>>>(qkv_w, qkv_wt, HS, 3 * HS);
    n = (long long)HS * HS;
    convert_wt_kernel<<<cdiv((unsigned)n, 256), 256, 0, stream>>>(proj_w, proj_wt, HS, HS);
    n = (long long)HS * MLP;
    convert_wt_kernel<<<cdiv((unsigned)n, 256), 256, 0, stream>>>(fc1_w, fc1_wt, HS, MLP);
    n = (long long)MLP * HS;
    convert_wt_kernel<<<cdiv((unsigned)n, 256), 256, 0, stream>>>(fc2_w, fc2_wt, MLP, HS);
  }

  // 2) LN1 + window partition -> xw (f16, zero-padded windows)
  ln1_window_kernel<<<MTOK, 256, 0, stream>>>(x, ln1_g, ln1_b, xw);

  // 3) QKV GEMM [19600 x 768] x [768 x 2304] -> scatter q/k/vT (f16)
  {
    EpiQKV epi{qkv_b, qh, kh, vh};
    dim3 grid(3 * HS / 64, cdiv(MTOK, 64));
    gemm_wmma_kernel<EpiQKV><<<grid, 128, 0, stream>>>(xw, qkv_wt, MTOK, 3 * HS, HS, epi);
  }

  // 4) decomposed rel-pos dot products
  {
    long long n = (long long)NWIN * NH * S * WSZ;
    relpos_kernel<<<cdiv((unsigned)n, 256), 256, 0, stream>>>(qh, rph, rpw, relh, relw);
  }

  // 5) windowed attention (softmax(qk^T/8 + bias) @ v), WMMA both GEMMs
  attn_kernel<<<NWIN * NH * STIL, 32, 0, stream>>>(qh, kh, vh, relh, relw, attn_h);

  // 6) proj GEMM + bias + unpartition + residual -> x2 (fp32)
  {
    EpiProj epi{proj_b, x, x2};
    dim3 grid(HS / 64, cdiv(MTOK, 64));
    gemm_wmma_kernel<EpiProj><<<grid, 128, 0, stream>>>(attn_h, proj_wt, MTOK, HS, HS, epi);
  }

  // 7) LN2 -> y (f16)
  ln2_kernel<<<MFLAT, 256, 0, stream>>>(x2, ln2_g, ln2_b, y);

  // 8) FC1 GEMM + GELU -> h1 (f16)
  {
    EpiGelu epi{fc1_b, h1};
    dim3 grid(MLP / 64, cdiv(MFLAT, 64));
    gemm_wmma_kernel<EpiGelu><<<grid, 128, 0, stream>>>(y, fc1_wt, MFLAT, MLP, HS, epi);
  }

  // 9) FC2 GEMM + bias + residual -> out (fp32)
  {
    EpiFC2 epi{fc2_b, x2, out};
    dim3 grid(HS / 64, cdiv(MFLAT, 64));
    gemm_wmma_kernel<EpiFC2><<<grid, 128, 0, stream>>>(h1, fc2_wt, MFLAT, HS, MLP, epi);
  }
}
